// GTLayer_86028194939205
// MI455X (gfx1250) — matrix operand order repaired and checked by reading, not compiled, not run on hardware
//
#include <hip/hip_runtime.h>

// ---------------------------------------------------------------------------
// MI455X (gfx1250, wave32) implementation of the GT layer.
// Matrix path: v_wmma_f32_16x16x32_bf16 (bf16 A/B, fp32 accum).
// Tile staging: CDNA5 async global->LDS copies (ASYNCcnt) where contiguous.
// ---------------------------------------------------------------------------

typedef __bf16 bf16;
typedef bf16  bf16x8  __attribute__((ext_vector_type(8)));
typedef bf16  bf16x16 __attribute__((ext_vector_type(16)));
typedef float f32x8   __attribute__((ext_vector_type(8)));

#define BB 2
#define SS 2048
#define DD 1024
#define HH 16
#define HD 64
#define S2 4096
#define KC 128
#define LN_EPS 1e-5f
#define NEG_E  -1e-12f

static __device__ __forceinline__ f32x8 wmma_bf16(bf16x16 a, bf16x16 b, f32x8 c) {
  return __builtin_amdgcn_wmma_f32_16x16x32_bf16(false, a, false, b, (short)0, c,
                                                 false, false);
}
static __device__ __forceinline__ bf16x16 cat8(bf16x8 lo, bf16x8 hi) {
  return __builtin_shufflevector(lo, hi, 0, 1, 2, 3, 4, 5, 6, 7,
                                         8, 9, 10, 11, 12, 13, 14, 15);
}
static __device__ __forceinline__ f32x8 zero8() {
  f32x8 z;
#pragma unroll
  for (int i = 0; i < 8; i++) z[i] = 0.0f;
  return z;
}

// CDNA5 async DMA: 16B per lane, global -> LDS, tracked by ASYNCcnt.
// ISA: GLOBAL_LOAD_ASYNC_TO_LDS_B128 VDST(LDS byte addr), VADDR(64b), off.
// Generic pointers to __shared__ carry the LDS offset in their low 32 bits.
static __device__ __forceinline__ void async_load_b128(const void* gptr,
                                                       const void* lptr) {
  asm volatile("global_load_async_to_lds_b128 %0, %1, off"
               :
               : "v"((unsigned)(size_t)lptr), "v"(gptr)
               : "memory");
}
static __device__ __forceinline__ void wait_async0() {
  asm volatile("s_wait_asynccnt 0x0" ::: "memory");
}

// ---------------------------------------------------------------------------
// Tiled bf16 WMMA GEMM: C[M,N] = A[M,K] @ B, optional bias / leaky epilogue.
// A row-major [M,K]. If bT==0: B row-major [K,N]; if bT==1: B is [N,K].
// 256 threads = 8 waves; WG tile 128x128, wave tile 64x32 (4x2 WMMA tiles).
// M,K multiples of 128/64; N multiple of 128 (or ==128 for the VQ GEMM).
// ---------------------------------------------------------------------------
__global__ __launch_bounds__(256) void gemm_bf16_kernel(
    const bf16* __restrict__ A, const bf16* __restrict__ Bm,
    float* __restrict__ Cf, bf16* __restrict__ Cb,
    const float* __restrict__ bias,
    int M, int N, int Kd, int leaky, int bT) {
  __shared__ __attribute__((aligned(16))) bf16 lA[128][72];  // [m][k]
  __shared__ __attribute__((aligned(16))) bf16 lB[128][72];  // [n][k] (transposed tile)

  const int tid  = threadIdx.x;
  const int lane = tid & 31;
  const int w    = tid >> 5;
  const int wm   = w & 1;    // 2 waves along M
  const int wn   = w >> 1;   // 4 waves along N
  const int g    = lane >> 4;
  const int l16  = lane & 15;
  const int mBase = blockIdx.y * 128;
  const int nBase = blockIdx.x * 128;

  f32x8 acc[4][2];
#pragma unroll
  for (int mt = 0; mt < 4; mt++)
#pragma unroll
    for (int nt = 0; nt < 2; nt++) acc[mt][nt] = zero8();

  for (int ks = 0; ks < Kd; ks += 64) {
    // CDNA5 speculative prefetch of the next K-stage (global_prefetch_b8).
    if (ks + 64 < Kd && tid < 128)
      __builtin_prefetch(&A[(size_t)(mBase + tid) * Kd + ks + 64], 0, 3);

    __syncthreads();
    // Stage A tile [128 x 64] via async global->LDS b128 DMA (ASYNCcnt).
#pragma unroll
    for (int i = 0; i < 4; i++) {
      int idx = tid + i * 256;
      int row = idx >> 3;
      int cq  = idx & 7;
      async_load_b128(&A[(size_t)(mBase + row) * Kd + ks + cq * 8],
                      &lA[row][cq * 8]);
    }
    // Stage B tile transposed into lB[n][k].
    if (!bT) {
#pragma unroll
      for (int i = 0; i < 4; i++) {
        int idx = tid + i * 256;
        int n  = idx & 127;
        int kq = idx >> 7;
#pragma unroll
        for (int j = 0; j < 8; j++)
          lB[n][kq * 8 + j] = Bm[(size_t)(ks + kq * 8 + j) * N + nBase + n];
      }
    } else {  // B already [N][K]: straight async b128 copies
#pragma unroll
      for (int i = 0; i < 4; i++) {
        int idx = tid + i * 256;
        int n  = idx & 127;
        int kq = idx >> 7;
        async_load_b128(&Bm[(size_t)(nBase + n) * Kd + ks + kq * 8],
                        &lB[n][kq * 8]);
      }
    }
    wait_async0();
    __syncthreads();

#pragma unroll
    for (int kk = 0; kk < 64; kk += 32) {
      bf16x16 af[4];
#pragma unroll
      for (int mt = 0; mt < 4; mt++) {
        int r = wm * 64 + mt * 16 + l16;  // A layout: lane -> row M
        bf16x8 lo = *(const bf16x8*)&lA[r][kk + g * 8];        // K = g*8+0..7
        bf16x8 hi = *(const bf16x8*)&lA[r][kk + 16 + g * 8];   // K = 16+g*8+0..7
        af[mt] = cat8(lo, hi);
      }
      bf16x16 bfr[2];
#pragma unroll
      for (int nt = 0; nt < 2; nt++) {
        int c = wn * 32 + nt * 16 + l16;  // B layout: lane -> col N
        bf16x8 lo = *(const bf16x8*)&lB[c][kk + g * 8];        // K = g*8+0..7
        bf16x8 hi = *(const bf16x8*)&lB[c][kk + 16 + g * 8];   // K = 16+g*8+0..7
        bfr[nt] = cat8(lo, hi);
      }
#pragma unroll
      for (int mt = 0; mt < 4; mt++)
#pragma unroll
        for (int nt = 0; nt < 2; nt++)
          acc[mt][nt] = wmma_bf16(af[mt], bfr[nt], acc[mt][nt]);
    }
  }

  // Epilogue: C layout row = g*8+e, col = l16.
#pragma unroll
  for (int mt = 0; mt < 4; mt++)
#pragma unroll
    for (int nt = 0; nt < 2; nt++)
#pragma unroll
      for (int e = 0; e < 8; e++) {
        int gr = mBase + wm * 64 + mt * 16 + g * 8 + e;
        int gc = nBase + wn * 32 + nt * 16 + l16;
        float v = acc[mt][nt][e];
        if (bias) v += bias[gc];
        if (leaky) v = (v >= 0.0f) ? v : 0.2f * v;
        if (Cf) Cf[(size_t)gr * N + gc] = v;
        if (Cb) Cb[(size_t)gr * N + gc] = (bf16)v;
      }
}

// ---------------------------------------------------------------------------
// Flash-style masked attention. One WG = 4 waves, each wave owns 16 q rows.
// Qp: [B,S,D] bf16 (head h = columns h*64..); K2: [B,H,2S,64]; V2T: [B,H,64,2S].
// mask: int32 [B,1,S,2S]. ctx: [B,S,D] bf16.
// ---------------------------------------------------------------------------
__global__ __launch_bounds__(128) void flash_attn_kernel(
    const bf16* __restrict__ Qp, const bf16* __restrict__ K2,
    const bf16* __restrict__ V2T, const int* __restrict__ mask,
    bf16* __restrict__ ctx) {
  __shared__ __attribute__((aligned(16))) float sP[4][16][33];
  __shared__ __attribute__((aligned(16))) bf16  sW[4][16][40];
  __shared__ float sM[4][16], sL[4][16], sC[4][16];

  const int tid  = threadIdx.x;
  const int lane = tid & 31;
  const int w    = tid >> 5;
  const int g    = lane >> 4;
  const int l16  = lane & 15;
  const int bh   = blockIdx.y;
  const int b    = bh / HH;
  const int hh   = bh % HH;
  const int qBase = blockIdx.x * 64 + w * 16;

  // Q A-fragments (16 rows x 64 dims -> two 16x32 k-steps), loaded once.
  bf16x16 aq[2];
  {
    const bf16* qrow = Qp + (size_t)(b * SS + qBase + l16) * DD + hh * HD;
#pragma unroll
    for (int kkI = 0; kkI < 2; kkI++) {
      bf16x8 lo = *(const bf16x8*)(qrow + kkI * 32 + g * 8);
      bf16x8 hi = *(const bf16x8*)(qrow + kkI * 32 + 16 + g * 8);
      aq[kkI] = cat8(lo, hi);
    }
  }
  if (lane < 16) { sM[w][lane] = -1e30f; sL[w][lane] = 0.0f; }
  __syncthreads();

  f32x8 cacc[4];
#pragma unroll
  for (int ct = 0; ct < 4; ct++) cacc[ct] = zero8();

  const bf16* Kb   = K2  + (size_t)bh * S2 * HD;
  const bf16* Vb   = V2T + (size_t)bh * HD * S2;
  const int*  mrow = mask + (size_t)b * SS * S2;

  for (int kc = 0; kc < S2; kc += 32) {
    // scores[16 q x 32 keys] = Q @ K2^T
    f32x8 sa[2];
#pragma unroll
    for (int nt = 0; nt < 2; nt++) {
      sa[nt] = zero8();
      int key = kc + nt * 16 + l16;
#pragma unroll
      for (int kkI = 0; kkI < 2; kkI++) {
        const bf16* kp = Kb + (size_t)key * HD + kkI * 32 + g * 8;
        bf16x16 bk = cat8(*(const bf16x8*)kp, *(const bf16x8*)(kp + 16));
        sa[nt] = wmma_bf16(aq[kkI], bk, sa[nt]);
      }
    }
    // scale 1/sqrt(64), masked_fill(-1e-12), spill to LDS.
#pragma unroll
    for (int nt = 0; nt < 2; nt++) {
      int key = kc + nt * 16 + l16;
#pragma unroll
      for (int e = 0; e < 8; e++) {
        int row = g * 8 + e;
        float v = sa[nt][e] * 0.125f;
        if (mrow[(size_t)(qBase + row) * S2 + key] == 0) v = NEG_E;
        sP[w][row][nt * 16 + l16] = v;
      }
    }
    __syncthreads();
    // Online-softmax stats: lane r (<16) owns row r.
    if (lane < 16) {
      int row = lane;
      float mold = sM[w][row], lold = sL[w][row];
      float mc = -1e30f;
#pragma unroll
      for (int j = 0; j < 32; j++) mc = fmaxf(mc, sP[w][row][j]);
      float mnew = fmaxf(mold, mc);
      float corr = __expf(mold - mnew);
      float s = 0.0f;
#pragma unroll
      for (int j = 0; j < 32; j++) {
        float e = __expf(sP[w][row][j] - mnew);
        sW[w][row][j] = (bf16)e;
        s += e;
      }
      sM[w][row] = mnew;
      sL[w][row] = lold * corr + s;
      sC[w][row] = corr;
    }
    __syncthreads();
    // ctx = corr*ctx + P @ V   (P reloaded from LDS in A-operand layout)
    bf16x16 aw;
    {
      bf16x8 lo = *(const bf16x8*)&sW[w][l16][g * 8];
      bf16x8 hi = *(const bf16x8*)&sW[w][l16][16 + g * 8];
      aw = cat8(lo, hi);
    }
#pragma unroll
    for (int ct = 0; ct < 4; ct++) {
#pragma unroll
      for (int e = 0; e < 8; e++) cacc[ct][e] *= sC[w][g * 8 + e];
      int dim = ct * 16 + l16;
      const bf16* vp = Vb + (size_t)dim * S2 + kc + g * 8;
      bf16x16 bv = cat8(*(const bf16x8*)vp, *(const bf16x8*)(vp + 16));
      cacc[ct] = wmma_bf16(aw, bv, cacc[ct]);
    }
    __syncthreads();
  }

  // Normalize and scatter back to [B,S,H,HD] = [B,S,D].
#pragma unroll
  for (int ct = 0; ct < 4; ct++)
#pragma unroll
    for (int e = 0; e < 8; e++) {
      int row  = g * 8 + e;
      float nv = cacc[ct][e] / sL[w][row];
      ctx[(size_t)(b * SS + qBase + row) * DD + hh * HD + ct * 16 + l16] = (bf16)nv;
    }
}

// ---------------------------------------------------------------------------
// Elementwise / reduction helper kernels
// ---------------------------------------------------------------------------
__global__ void cvt_kernel(const float* __restrict__ in, bf16* __restrict__ out, int n) {
  int i = blockIdx.x * blockDim.x + threadIdx.x;
  int stride = gridDim.x * blockDim.x;
  for (; i < n; i += stride) out[i] = (bf16)in[i];
}

__global__ __launch_bounds__(256) void rownorm2_kernel(
    const float* __restrict__ x, float* __restrict__ out, int cols) {
  __shared__ float red[256];
  int row = blockIdx.x, t = threadIdx.x;
  float s = 0.0f;
  for (int c = t; c < cols; c += 256) {
    float v = x[(size_t)row * cols + c];
    s += v * v;
  }
  red[t] = s; __syncthreads();
  for (int o = 128; o > 0; o >>= 1) {
    if (t < o) red[t] += red[t + o];
    __syncthreads();
  }
  if (t == 0) out[row] = red[0];
}

// softmax(-d2) @ codebook + per-row squared error (VQ straight-through value).
__global__ __launch_bounds__(128) void vq_kernel(
    const float* __restrict__ dot, const float* __restrict__ xx,
    const float* __restrict__ cbn, const bf16* __restrict__ cb,
    const float* __restrict__ x, bf16* __restrict__ hnew,
    float* __restrict__ partial) {
  __shared__ float swv[KC];
  __shared__ float red[128];
  int row = blockIdx.x, t = threadIdx.x;
  float s = -(xx[row] - 2.0f * dot[(size_t)row * KC + t] + cbn[t]);
  red[t] = s; __syncthreads();
  for (int o = 64; o > 0; o >>= 1) {
    if (t < o) red[t] = fmaxf(red[t], red[t + o]);
    __syncthreads();
  }
  float mx = red[0]; __syncthreads();
  float e = __expf(s - mx);
  red[t] = e; __syncthreads();
  for (int o = 64; o > 0; o >>= 1) {
    if (t < o) red[t] += red[t + o];
    __syncthreads();
  }
  float inv = 1.0f / red[0]; __syncthreads();
  swv[t] = e * inv;
  __syncthreads();
  float lsq = 0.0f;
  for (int d = t; d < DD; d += 128) {
    float q = 0.0f;
    for (int n = 0; n < KC; n++) q += swv[n] * (float)cb[(size_t)n * DD + d];
    float diff = q - x[(size_t)row * DD + d];
    lsq += diff * diff;
    hnew[(size_t)row * DD + d] = (bf16)q;
  }
  red[t] = lsq; __syncthreads();
  for (int o = 64; o > 0; o >>= 1) {
    if (t < o) red[t] += red[t + o];
    __syncthreads();
  }
  if (t == 0) partial[row] = red[0];
}

__global__ __launch_bounds__(256) void loss_kernel(
    const float* __restrict__ partial, const float* __restrict__ cc,
    float* __restrict__ out) {
  __shared__ float red[256];
  int t = threadIdx.x;
  float s = 0.0f;
  for (int i = t; i < BB * SS; i += 256) s += partial[i];
  red[t] = s; __syncthreads();
  for (int o = 128; o > 0; o >>= 1) {
    if (t < o) red[t] += red[t + o];
    __syncthreads();
  }
  if (t == 0) out[0] = red[0] * (1.0f + cc[0]) / (float)((size_t)BB * SS * DD);
}

// K2[b][h][s2][d] from the two K projection halves.
__global__ void build_k2_kernel(const bf16* __restrict__ k1,
                                const bf16* __restrict__ k2h,
                                bf16* __restrict__ K2) {
  size_t total = (size_t)BB * HH * S2 * HD;
  size_t stride = (size_t)gridDim.x * blockDim.x;
  for (size_t i = (size_t)blockIdx.x * blockDim.x + threadIdx.x; i < total; i += stride) {
    int d = (int)(i % HD);
    size_t r = i / HD;
    int s2 = (int)(r % S2); r /= S2;
    int h = (int)(r % HH);
    int b = (int)(r / HH);
    bf16 v = (s2 < SS) ? k1[(size_t)(b * SS + s2) * DD + h * HD + d]
                       : k2h[(size_t)(b * SS + s2 - SS) * DD + h * HD + d];
    K2[i] = v;
  }
}

// V2T[b][h][d][s2] (transposed so attention B-fragments are contiguous).
__global__ void build_v2t_kernel(const bf16* __restrict__ v1,
                                 const bf16* __restrict__ v2h,
                                 bf16* __restrict__ V2T) {
  size_t total = (size_t)BB * HH * HD * S2;
  size_t stride = (size_t)gridDim.x * blockDim.x;
  for (size_t i = (size_t)blockIdx.x * blockDim.x + threadIdx.x; i < total; i += stride) {
    int s2 = (int)(i % S2);
    size_t r = i / S2;
    int d = (int)(r % HD); r /= HD;
    int h = (int)(r % HH);
    int b = (int)(r / HH);
    bf16 v = (s2 < SS) ? v1[(size_t)(b * SS + s2) * DD + h * HD + d]
                       : v2h[(size_t)(b * SS + s2 - SS) * DD + h * HD + d];
    V2T[i] = v;
  }
}

// y = LayerNorm(a + b) * gamma + beta ; fp32 out (+ optional bf16 copy).
__global__ __launch_bounds__(256) void ln_kernel(
    const float* __restrict__ a, const float* __restrict__ b,
    const float* __restrict__ gamma, const float* __restrict__ beta,
    float* __restrict__ outF, bf16* __restrict__ outB) {
  __shared__ float rs_[256], rq[256];
  __shared__ float mbuf[2];
  int row = blockIdx.x, t = threadIdx.x;
  float s = 0.0f, q = 0.0f;
  float vals[4];
#pragma unroll
  for (int i = 0; i < 4; i++) {
    int c = t + i * 256;
    float v = a[(size_t)row * DD + c] + b[(size_t)row * DD + c];
    vals[i] = v; s += v; q += v * v;
  }
  rs_[t] = s; rq[t] = q; __syncthreads();
  for (int o = 128; o > 0; o >>= 1) {
    if (t < o) { rs_[t] += rs_[t + o]; rq[t] += rq[t + o]; }
    __syncthreads();
  }
  if (t == 0) {
    float m = rs_[0] / (float)DD;
    float var = rq[0] / (float)DD - m * m;
    mbuf[0] = m; mbuf[1] = rsqrtf(var + LN_EPS);
  }
  __syncthreads();
  float m = mbuf[0], r = mbuf[1];
#pragma unroll
  for (int i = 0; i < 4; i++) {
    int c = t + i * 256;
    float y = (vals[i] - m) * r * gamma[c] + beta[c];
    outF[(size_t)row * DD + c] = y;
    if (outB) outB[(size_t)row * DD + c] = (bf16)y;
  }
}

// ---------------------------------------------------------------------------
extern "C" void kernel_launch(void* const* d_in, const int* in_sizes, int n_in,
                              void* d_out, int out_size, void* d_ws, size_t ws_size,
                              hipStream_t stream) {
  const float* h    = (const float*)d_in[0];
  const int*   mask = (const int*)d_in[1];
  const float* Wq   = (const float*)d_in[2];
  const float* Wk   = (const float*)d_in[3];
  const float* Wv   = (const float*)d_in[4];
  const float* Wo   = (const float*)d_in[5];
  const float* W1   = (const float*)d_in[6];
  const float* b1   = (const float*)d_in[7];
  const float* W2   = (const float*)d_in[8];
  const float* b2   = (const float*)d_in[9];
  const float* ln1g = (const float*)d_in[10];
  const float* ln1b = (const float*)d_in[11];
  const float* ln2g = (const float*)d_in[12];
  const float* ln2b = (const float*)d_in[13];
  const float* cbk  = (const float*)d_in[14];
  const float* cc   = (const float*)d_in[15];
  float* out = (float*)d_out;

  char* ws = (char*)d_ws;
  size_t off = 0;
  auto alloc = [&](size_t bytes) -> void* {
    void* p = ws + off;
    off += (bytes + 255) & ~(size_t)255;
    return p;
  };

  const int nBS  = BB * SS;        // 4096 rows
  const int nBSD = BB * SS * DD;   // 4194304 elements

  bf16* h_bf  = (bf16*)alloc((size_t)nBSD * 2);
  bf16* Wq_bf = (bf16*)alloc((size_t)DD * DD * 2);
  bf16* Wk_bf = (bf16*)alloc((size_t)DD * DD * 2);
  bf16* Wv_bf = (bf16*)alloc((size_t)DD * DD * 2);
  bf16* Wo_bf = (bf16*)alloc((size_t)DD * DD * 2);
  bf16* W1_bf = (bf16*)alloc((size_t)DD * DD * 2);
  bf16* W2_bf = (bf16*)alloc((size_t)DD * DD * 2);
  bf16* cb_bf = (bf16*)alloc((size_t)KC * DD * 2);
  bf16* Qp    = (bf16*)alloc((size_t)nBSD * 2);
  bf16* k1    = (bf16*)alloc((size_t)nBSD * 2);
  bf16* v1    = (bf16*)alloc((size_t)nBSD * 2);
  float* dotb = (float*)alloc((size_t)nBS * KC * 4);
  float* xx   = (float*)alloc((size_t)nBS * 4);
  float* cbn  = (float*)alloc((size_t)KC * 4);
  bf16* hnew  = (bf16*)alloc((size_t)nBSD * 2);
  float* part = (float*)alloc((size_t)nBS * 4);
  bf16* k2h   = (bf16*)alloc((size_t)nBSD * 2);
  bf16* v2h   = (bf16*)alloc((size_t)nBSD * 2);
  bf16* K2    = (bf16*)alloc((size_t)BB * HH * S2 * HD * 2);
  bf16* V2T   = (bf16*)alloc((size_t)BB * HH * HD * S2 * 2);
  bf16* ctx   = (bf16*)alloc((size_t)nBSD * 2);
  float* hsa  = (float*)alloc((size_t)nBSD * 4);
  float* h1f  = (float*)alloc((size_t)nBSD * 4);
  bf16* h1b   = (bf16*)alloc((size_t)nBSD * 2);
  bf16* t1    = (bf16*)alloc((size_t)nBSD * 2);
  float* t2   = (float*)alloc((size_t)nBSD * 4);

  // fp32 -> bf16 conversions
  cvt_kernel<<<1024, 256, 0, stream>>>(h, h_bf, nBSD);
  cvt_kernel<<<512, 256, 0, stream>>>(Wq, Wq_bf, DD * DD);
  cvt_kernel<<<512, 256, 0, stream>>>(Wk, Wk_bf, DD * DD);
  cvt_kernel<<<512, 256, 0, stream>>>(Wv, Wv_bf, DD * DD);
  cvt_kernel<<<512, 256, 0, stream>>>(Wo, Wo_bf, DD * DD);
  cvt_kernel<<<512, 256, 0, stream>>>(W1, W1_bf, DD * DD);
  cvt_kernel<<<512, 256, 0, stream>>>(W2, W2_bf, DD * DD);
  cvt_kernel<<<128, 256, 0, stream>>>(cbk, cb_bf, KC * DD);

  // VQ norms
  rownorm2_kernel<<<nBS, 256, 0, stream>>>(h, xx, DD);
  rownorm2_kernel<<<KC, 256, 0, stream>>>(cbk, cbn, DD);

  dim3 gD(DD / 128, nBS / 128);  // (8, 32)
  // Q/K/V projections of h
  gemm_bf16_kernel<<<gD, 256, 0, stream>>>(h_bf, Wq_bf, nullptr, Qp, nullptr, nBS, DD, DD, 0, 0);
  gemm_bf16_kernel<<<gD, 256, 0, stream>>>(h_bf, Wk_bf, nullptr, k1, nullptr, nBS, DD, DD, 0, 0);
  gemm_bf16_kernel<<<gD, 256, 0, stream>>>(h_bf, Wv_bf, nullptr, v1, nullptr, nBS, DD, DD, 0, 0);

  // VQ distance dot products: h @ codebook^T (codebook is [N][K] -> bT=1)
  dim3 gDot(1, nBS / 128);
  gemm_bf16_kernel<<<gDot, 256, 0, stream>>>(h_bf, cb_bf, dotb, nullptr, nullptr, nBS, KC, DD, 0, 1);

  // softmax-VQ + loss
  vq_kernel<<<nBS, 128, 0, stream>>>(dotb, xx, cbn, cb_bf, h, hnew, part);
  loss_kernel<<<1, 256, 0, stream>>>(part, cc, out + nBSD);

  // K/V projections of quantized stream
  gemm_bf16_kernel<<<gD, 256, 0, stream>>>(hnew, Wk_bf, nullptr, k2h, nullptr, nBS, DD, DD, 0, 0);
  gemm_bf16_kernel<<<gD, 256, 0, stream>>>(hnew, Wv_bf, nullptr, v2h, nullptr, nBS, DD, DD, 0, 0);

  // Assemble attention operands
  build_k2_kernel<<<2048, 256, 0, stream>>>(k1, k2h, K2);
  build_v2t_kernel<<<2048, 256, 0, stream>>>(v1, v2h, V2T);

  // Masked flash attention
  dim3 gF(SS / 64, BB * HH);  // (32, 32)
  flash_attn_kernel<<<gF, 128, 0, stream>>>(Qp, K2, V2T, mask, ctx);

  // Output projection + leaky, LN1, FFN, LN2
  gemm_bf16_kernel<<<gD, 256, 0, stream>>>(ctx, Wo_bf, hsa, nullptr, nullptr, nBS, DD, DD, 1, 0);
  ln_kernel<<<nBS, 256, 0, stream>>>(hsa, h, ln1g, ln1b, h1f, h1b);
  gemm_bf16_kernel<<<gD, 256, 0, stream>>>(h1b, W1_bf, nullptr, t1, b1, nBS, DD, DD, 1, 0);
  gemm_bf16_kernel<<<gD, 256, 0, stream>>>(t1, W2_bf, t2, nullptr, b2, nBS, DD, DD, 0, 0);
  ln_kernel<<<nBS, 256, 0, stream>>>(h1f, t2, ln2g, ln2b, out, nullptr);
}
